// SpinalCordCircuit_230119_15006615734208
// MI455X (gfx1250) — matrix operand order repaired and checked by reading, not compiled, not run on hardware
//
#include <hip/hip_runtime.h>

// ---------------------------------------------------------------------------
// MI455X (gfx1250) implementation of the SpinalCordCircuit scan.
//
// Design (latency-bound serial scan, B=256 independent rows):
//   * 16 workgroups, one per 16-row WMMA M-tile; 256 threads = 8 wave32.
//   * All six recurrent states + all small params live in LDS for the scan.
//   * GEMMs via v_wmma_f32_16x16x32_bf16 (bf16 operands, f32 accumulate).
//   * Prologue kernels pre-pack weights (~0.8MB, L2-resident) and afferents
//     (step-major, 64MB, streamed non-temporally) into WMMA fragment layouts.
//   * Global loads forced to address_space(1) -> global_load_b128 (no flat).
//   * Workspace requirement: >= 68 MB.
// ---------------------------------------------------------------------------

typedef __attribute__((ext_vector_type(16))) __bf16 v16bf;
typedef __attribute__((ext_vector_type(8)))  float  v8f;

#define EPSF 1e-5f

#if defined(__HIP_DEVICE_COMPILE__)
#define GAS __attribute__((address_space(1)))
#else
#define GAS
#endif

// ws layout (bytes)
#define OFF_H2H   0u          // 6 * 64 tiles * 1KB  = 393216 B
#define OFF_X2H   393216u     // 6 * 32 tiles * 1KB  = 196608 B
#define OFF_CONV  589824u     // 6 * 32 tiles * 1KB  = 196608 B
#define OFF_CVEX  786432u     // 2 * 16 tiles * 1KB  =  32768 B
#define OFF_AFF   1048576u    // 256*16*4*4 tiles * 1KB = 64 MB

// LDS parameter block offsets (floats)
#define PAR_H2LW  0
#define PAR_H2LB  1536
#define PAR_XLW   3072
#define PAR_XLB   3840
#define PAR_NG    4608
#define PAR_NB    6912
#define PAR_AG    9216
#define PAR_AB    11520
#define PAR_CONVB 13824
#define PAR_CVEXB 15360
#define PAR_TOT   15616

__device__ __forceinline__ unsigned short f2bf(float f) {
  union { float f; unsigned u; } c; c.f = f;
  unsigned u = c.u;
  return (unsigned short)((u + 0x7FFFu + ((u >> 16) & 1u)) >> 16); // RNE
}

// sum over the 16-lane half-group owning one matrix row
__device__ __forceinline__ float redsum16(float v) {
  v += __shfl_xor(v, 8, 32);
  v += __shfl_xor(v, 4, 32);
  v += __shfl_xor(v, 2, 32);
  v += __shfl_xor(v, 1, 32);
  return v;
}

// ---- fragment loads with explicit address spaces -------------------------
__device__ __forceinline__ v16bf ldl_frag(const unsigned short* p) {   // LDS
  return *(const v16bf*)p;
}
__device__ __forceinline__ v16bf ldg_frag(const unsigned short* p) {   // global, RT
  const GAS v16bf* gp = (const GAS v16bf*)(unsigned long long)p;
  return *gp;
}
__device__ __forceinline__ v16bf ldg_frag_nt(const unsigned short* p) { // global, NT
  const GAS v16bf* gp = (const GAS v16bf*)(unsigned long long)p;
  return __builtin_nontemporal_load(gp);
}
__device__ __forceinline__ void stg_nt(float* p, float v) {             // global NT store
  GAS float* gp = (GAS float*)(unsigned long long)p;
  __builtin_nontemporal_store(v, gp);
}

// A from LDS, B from global (weights, L2-resident)
__device__ __forceinline__ v8f wmma_lg(const unsigned short* a_lds,
                                       const unsigned short* b_glb, v8f acc) {
  return __builtin_amdgcn_wmma_f32_16x16x32_bf16(false, ldl_frag(a_lds),
                                                 false, ldg_frag(b_glb),
                                                 (short)0, acc, false, false);
}
// A from global (streamed afferents, NT), B from global (weights)
__device__ __forceinline__ v8f wmma_gg(const unsigned short* a_glb,
                                       const unsigned short* b_glb, v8f acc) {
  return __builtin_amdgcn_wmma_f32_16x16x32_bf16(false, ldg_frag_nt(a_glb),
                                                 false, ldg_frag(b_glb),
                                                 (short)0, acc, false, false);
}

// scatter one f32 activation element (m,k) into bf16 A-fragment layout
// A 16-bit 16x32 tile: lane = m | (hi<<4); K = (e&7) + 8*hi + 16*(e>>3)
__device__ __forceinline__ void frag_write(unsigned short* frag, int m, int k, float v) {
  int kt = k >> 5, kk = k & 31;
  int hi = (kk >> 3) & 1;
  int e  = (kk & 7) | ((kk >> 4) << 3);
  frag[kt * 512 + ((m | (hi << 4)) << 4) + e] = f2bf(v);
}

// ---------------------------------------------------------------------------
// Prologue: pack weight matrix (L, NOUT, KIN) row-major into B-fragment tiles.
// B 16-bit 32x16 tile: lane = n | (hi<<4) with hi = (kk>>3)&1; e = (kk&7) + 8*(kk>>4).
// Tile order: [l][kt][nt], 512 bf16 per tile.
// ---------------------------------------------------------------------------
__global__ void pack_w(const float* __restrict__ src, unsigned short* __restrict__ dst,
                       int KIN, int NOUT, int L) {
  int idx = blockIdx.x * 256 + threadIdx.x;
  int total = L * NOUT * KIN;
  if (idx >= total) return;
  int k = idx % KIN;
  int rem = idx / KIN;
  int n = rem % NOUT;
  int l = rem / NOUT;
  float v = src[(size_t)(l * NOUT + n) * KIN + k];
  int NT = NOUT >> 4;
  int nt = n >> 4, nn = n & 15;
  int kt = k >> 5, kk = k & 31;
  int hi = (kk >> 3) & 1, e = (kk & 7) | ((kk >> 4) << 3);
  size_t lstride = (size_t)(KIN >> 5) * NT * 512;
  dst[(size_t)l * lstride + (size_t)(kt * NT + nt) * 512 + ((nn | (hi << 4)) << 4) + e] = f2bf(v);
}

// ---------------------------------------------------------------------------
// Prologue: repack afferents (B, 4N, T) f32 -> step-major bf16 A-fragments.
// dst tile order: [t][wg][slice][kt][lane][e]; slice: 0=Ia_f 1=Ia_e 2=II_f 3=II_e
// ---------------------------------------------------------------------------
__global__ void pack_aff(const float* __restrict__ aff, unsigned short* __restrict__ dst) {
  int bc = blockIdx.x;      // b*512 + c
  int t  = threadIdx.x;     // time (coalesced reads)
  int b = bc >> 9, c = bc & 511;
  int s = c >> 7, k = c & 127;
  int wg = b >> 4, m = b & 15;
  int kt = k >> 5, kk = k & 31;
  int hi = (kk >> 3) & 1;
  int e  = (kk & 7) | ((kk >> 4) << 3);
  float v = aff[(size_t)bc * 256 + t];
  size_t off = ((((size_t)(t * 16 + wg) * 4 + s) * 4 + kt) * 32 + (m | (hi << 4))) * 16 + e;
  dst[off] = f2bf(v);
}

// ---------------------------------------------------------------------------
// One recurrent gated layer (all 8 waves cooperate). Force-inlined so the
// address-space inference pass sees kernel-arg / LDS provenance everywhere.
// ---------------------------------------------------------------------------
template<int P, bool EI>
__device__ __forceinline__ void run_layer(
    int tid, int wg, int t,
    const float* xp0, int xst0, const float* xp1, int xst1,
    const float* xp2, int xst2,
    float* __restrict__ hx,            // [16*128] this layer's state (LDS)
    float* __restrict__ h2buf,         // [16*256] scratch (LDS)
    float* __restrict__ xbuf,          // [16*128] scratch (LDS)
    unsigned short* __restrict__ hxfrag,
    unsigned short* __restrict__ xfrag,
    const unsigned short* __restrict__ wh2h,  // 64 tiles (global)
    const unsigned short* __restrict__ wx2h,  // 32 tiles (global)
    const float* __restrict__ h2lw, const float* __restrict__ h2lb,  // LDS
    const float* __restrict__ xlw,  const float* __restrict__ xlb,   // LDS
    const float* __restrict__ ng,   const float* __restrict__ nb,    // LDS
    const float* __restrict__ ag,   const float* __restrict__ ab,    // LDS
    float* __restrict__ outp)
{
  const int wave = tid >> 5, lane = tid & 31;
  const int m = tid >> 4, kg = tid & 15;

  // L1: pack hx into A-fragments
#pragma unroll
  for (int j = 0; j < 8; ++j) {
    int k = kg + 16 * j;
    frag_write(hxfrag, m, k, hx[m * 128 + k]);
  }
  __syncthreads();

  // L2: h2 = hx @ h2h_W^T  (K=128 -> 4 ktiles, N=256 -> 16 ntiles)
#pragma unroll
  for (int i = 0; i < 2; ++i) {
    int nt = wave + 8 * i;
    v8f acc = {0.f,0.f,0.f,0.f,0.f,0.f,0.f,0.f};
#pragma unroll
    for (int kt = 0; kt < 4; ++kt)
      acc = wmma_lg(hxfrag + kt * 512 + lane * 16,
                    wh2h + (kt * 16 + nt) * 512 + lane * 16, acc);
    int n = lane & 15, mb = (lane >> 4) * 8;
#pragma unroll
    for (int r = 0; r < 8; ++r) h2buf[(mb + r) * 256 + nt * 16 + n] = acc[r];
  }
  __syncthreads();

  // L3: LN(h2) over 256; LN(xs) per group over 128; gates; s; x = relu(s+h_i)
  float s1 = 0.f, s2 = 0.f;
#pragma unroll
  for (int j = 0; j < 16; ++j) { float v = h2buf[m * 256 + kg + 16 * j]; s1 += v; s2 += v * v; }
  s1 = redsum16(s1); s2 = redsum16(s2);
  float h2m = s1 * (1.f / 256.f);
  float h2r = rsqrtf(s2 * (1.f / 256.f) - h2m * h2m + EPSF);

  const float* xp[3]   = { xp0, xp1, xp2 };
  const int    xstr[3] = { xst0, xst1, xst2 };
  float gm[P], gr[P];
#pragma unroll
  for (int pi = 0; pi < P; ++pi) {
    float a = 0.f, b = 0.f;
#pragma unroll
    for (int j = 0; j < 8; ++j) { float v = xp[pi][m * xstr[pi] + kg + 16 * j]; a += v; b += v * v; }
    a = redsum16(a); b = redsum16(b);
    gm[pi] = a * (1.f / 128.f);
    gr[pi] = rsqrtf(b * (1.f / 128.f) - gm[pi] * gm[pi] + EPSF);
  }
#pragma unroll
  for (int j = 0; j < 8; ++j) {
    int k = kg + 16 * j;
    float hiv = (h2buf[m * 256 + k]       - h2m) * h2r * h2lw[k]       + h2lb[k];
    float hgv = (h2buf[m * 256 + 128 + k] - h2m) * h2r * h2lw[128 + k] + h2lb[128 + k];
    h2buf[m * 256 + 128 + k] = hgv;          // keep h_g for L5
    float hxv = hx[m * 128 + k];
    float inl[P];
#pragma unroll
    for (int pi = 0; pi < P; ++pi) {
      float xv = xp[pi][m * xstr[pi] + k];
      float xn = (xv - gm[pi]) * gr[pi] * ng[pi * 128 + k] + nb[pi * 128 + k];
      float g  = 1.f / (1.f + __expf(-(ag[pi * 128 + k] * hxv + ab[pi * 128 + k])));
      inl[pi] = g * xn;
    }
    float s;
    if constexpr (EI) s = fmaxf(inl[0] + inl[1], 0.f) - fmaxf(inl[P - 1], 0.f);
    else              s = fmaxf(inl[0], 0.f);
    float x = fmaxf(s + hiv, 0.f);
    xbuf[m * 128 + k] = x;
    frag_write(xfrag, m, k, x);
  }
  __syncthreads();

  // L4: y = x @ x2h_W^T (K=128, N=128 -> 8 ntiles, one per wave); y -> h2buf[:, :128]
  {
    int nt = wave;
    v8f acc = {0.f,0.f,0.f,0.f,0.f,0.f,0.f,0.f};
#pragma unroll
    for (int kt = 0; kt < 4; ++kt)
      acc = wmma_lg(xfrag + kt * 512 + lane * 16,
                    wx2h + (kt * 8 + nt) * 512 + lane * 16, acc);
    int n = lane & 15, mb = (lane >> 4) * 8;
#pragma unroll
    for (int r = 0; r < 8; ++r) h2buf[(mb + r) * 256 + nt * 16 + n] = acc[r];
  }
  __syncthreads();

  // L5: LN(y) over 128; gg; state update (+ optional non-temporal output write)
  float a0 = 0.f, b0 = 0.f;
#pragma unroll
  for (int j = 0; j < 8; ++j) { float v = h2buf[m * 256 + kg + 16 * j]; a0 += v; b0 += v * v; }
  a0 = redsum16(a0); b0 = redsum16(b0);
  float ym = a0 * (1.f / 128.f);
  float yr = rsqrtf(b0 * (1.f / 128.f) - ym * ym + EPSF);
#pragma unroll
  for (int j = 0; j < 8; ++j) {
    int k = kg + 16 * j;
    float yn = (h2buf[m * 256 + k] - ym) * yr * xlw[k] + xlb[k];
    float gg = 1.f / (1.f + __expf(-(yn + h2buf[m * 256 + 128 + k])));
    float hn = (1.f - gg) * hx[m * 128 + k] + gg * xbuf[m * 128 + k];
    hx[m * 128 + k] = hn;
    if (outp) stg_nt(&outp[((size_t)(wg * 16 + m) * 256 + k) * 256 + t], hn);
  }
  __syncthreads();
}

// ---------------------------------------------------------------------------
// Main scan kernel: 16 blocks x 256 threads; full T=256 scan per block.
// ---------------------------------------------------------------------------
__global__ __launch_bounds__(256) void spinal_scan(
    const unsigned short* __restrict__ aff_pack,
    const unsigned short* __restrict__ wh2h_all,
    const unsigned short* __restrict__ wx2h_all,
    const unsigned short* __restrict__ wconv_all,
    const unsigned short* __restrict__ wcvex_all,
    const float* __restrict__ norm_g, const float* __restrict__ norm_b,
    const float* __restrict__ aff_g,  const float* __restrict__ aff_b,
    const float* __restrict__ x2h_lnw, const float* __restrict__ x2h_lnb,
    const float* __restrict__ h2h_lnw, const float* __restrict__ h2h_lnb,
    const float* __restrict__ convb,   const float* __restrict__ convexb,
    float* __restrict__ out)
{
  __shared__ float sm_states[6 * 16 * 128];   // exF exE IaiF IaiE mnF mnE
  __shared__ float sm_save[16 * 128];         // IaiF_old for the IaiE layer
  __shared__ float sm_cin[4 * 16 * 256];      // input gconv outputs
  __shared__ float sm_h2[16 * 256];
  __shared__ float sm_x[16 * 128];
  __shared__ float sm_g[4 * 16 * 128];        // exF2mn exE2mn IaiE2mnF IaiF2mnE
  __shared__ float sm_par[PAR_TOT];           // staged small parameters
  __shared__ __align__(32) unsigned short sm_hxfrag[4 * 512];
  __shared__ __align__(32) unsigned short sm_xfrag[4 * 512];

  const int tid = threadIdx.x, wg = blockIdx.x;
  const int wave = tid >> 5, lane = tid & 31;
  const int m = tid >> 4, kg = tid & 15;

  for (int i = tid; i < 6 * 16 * 128; i += 256) sm_states[i] = 0.f;
  // stage all small params into LDS once
  for (int i = tid; i < 1536; i += 256) {
    sm_par[PAR_H2LW + i] = h2h_lnw[i];
    sm_par[PAR_H2LB + i] = h2h_lnb[i];
    sm_par[PAR_CONVB + i] = convb[i];
  }
  for (int i = tid; i < 768; i += 256) {
    sm_par[PAR_XLW + i] = x2h_lnw[i];
    sm_par[PAR_XLB + i] = x2h_lnb[i];
  }
  for (int i = tid; i < 2304; i += 256) {
    sm_par[PAR_NG + i] = norm_g[i];
    sm_par[PAR_NB + i] = norm_b[i];
    sm_par[PAR_AG + i] = aff_g[i];
    sm_par[PAR_AB + i] = aff_b[i];
  }
  if (tid < 256) {
    sm_par[PAR_CVEXB + tid] = convexb[tid];
  }
  __syncthreads();

  for (int t = 0; t < 256; ++t) {
    const unsigned short* inbase = aff_pack + (size_t)(t * 16 + wg) * 8192;
    if (t + 1 < 256)
      __builtin_prefetch((const char*)(aff_pack + (size_t)((t + 1) * 16 + wg) * 8192) + tid * 64, 0, 1);

    // ---- S1: 4 input gconvs: c_in[j] = gconv(slice sIdx[j], convW[j]) -----
    // c_in order: 0:(iaF,convW0) 1:(iiF,convW1) 2:(iaE,convW2) 3:(iiE,convW3)
    {
      const int sIdx[4] = {0, 2, 1, 3};
      for (int idx = wave; idx < 64; idx += 8) {
        int j = idx >> 4, nt = idx & 15;
        const unsigned short* afr = inbase + sIdx[j] * 2048;
        int akt0 = (nt < 8) ? 0 : 2;          // first/second input half
        const unsigned short* bt = wconv_all + j * 16384;
        v8f acc = {0.f,0.f,0.f,0.f,0.f,0.f,0.f,0.f};
#pragma unroll
        for (int kt = 0; kt < 2; ++kt)
          acc = wmma_gg(afr + (akt0 + kt) * 512 + lane * 16,
                        bt + (kt * 16 + nt) * 512 + lane * 16, acc);
        int n = lane & 15, mb = (lane >> 4) * 8;
        float bias = sm_par[PAR_CONVB + j * 256 + nt * 16 + n];
#pragma unroll
        for (int r = 0; r < 8; ++r)
          sm_cin[j * 4096 + (mb + r) * 256 + nt * 16 + n] = acc[r] + bias;
      }
    }
    __syncthreads();

    // ---- exF (li=0), exE (li=1) ----
    run_layer<1, false>(tid, wg, t,
        &sm_cin[1 * 4096], 256, nullptr, 0, nullptr, 0,
        &sm_states[0 * 2048], sm_h2, sm_x, sm_hxfrag, sm_xfrag,
        wh2h_all + 0 * 32768, wx2h_all + 0 * 16384,
        &sm_par[PAR_H2LW + 0 * 256], &sm_par[PAR_H2LB + 0 * 256],
        &sm_par[PAR_XLW + 0 * 128],  &sm_par[PAR_XLB + 0 * 128],
        &sm_par[PAR_NG + 0 * 384],   &sm_par[PAR_NB + 0 * 384],
        &sm_par[PAR_AG + 0 * 384],   &sm_par[PAR_AB + 0 * 384], nullptr);
    run_layer<1, false>(tid, wg, t,
        &sm_cin[3 * 4096], 256, nullptr, 0, nullptr, 0,
        &sm_states[1 * 2048], sm_h2, sm_x, sm_hxfrag, sm_xfrag,
        wh2h_all + 1 * 32768, wx2h_all + 1 * 16384,
        &sm_par[PAR_H2LW + 1 * 256], &sm_par[PAR_H2LB + 1 * 256],
        &sm_par[PAR_XLW + 1 * 128],  &sm_par[PAR_XLB + 1 * 128],
        &sm_par[PAR_NG + 1 * 384],   &sm_par[PAR_NB + 1 * 384],
        &sm_par[PAR_AG + 1 * 384],   &sm_par[PAR_AB + 1 * 384], nullptr);

    // save IaiF old for the IaiE layer
    for (int i = tid; i < 2048; i += 256) sm_save[i] = sm_states[2 * 2048 + i];

    // ---- IaiF (li=2): xs = {IaF2Iai, IIF2Iai, IaiE_old} ----
    run_layer<3, true>(tid, wg, t,
        &sm_cin[0 * 4096 + 128], 256, &sm_cin[1 * 4096 + 128], 256,
        &sm_states[3 * 2048], 128,
        &sm_states[2 * 2048], sm_h2, sm_x, sm_hxfrag, sm_xfrag,
        wh2h_all + 2 * 32768, wx2h_all + 2 * 16384,
        &sm_par[PAR_H2LW + 2 * 256], &sm_par[PAR_H2LB + 2 * 256],
        &sm_par[PAR_XLW + 2 * 128],  &sm_par[PAR_XLB + 2 * 128],
        &sm_par[PAR_NG + 2 * 384],   &sm_par[PAR_NB + 2 * 384],
        &sm_par[PAR_AG + 2 * 384],   &sm_par[PAR_AB + 2 * 384], nullptr);
    // ---- IaiE (li=3): xs = {IaE2Iai, IIE2Iai, IaiF_old} ----
    run_layer<3, true>(tid, wg, t,
        &sm_cin[2 * 4096 + 128], 256, &sm_cin[3 * 4096 + 128], 256,
        sm_save, 128,
        &sm_states[3 * 2048], sm_h2, sm_x, sm_hxfrag, sm_xfrag,
        wh2h_all + 3 * 32768, wx2h_all + 3 * 16384,
        &sm_par[PAR_H2LW + 3 * 256], &sm_par[PAR_H2LB + 3 * 256],
        &sm_par[PAR_XLW + 3 * 128],  &sm_par[PAR_XLB + 3 * 128],
        &sm_par[PAR_NG + 3 * 384],   &sm_par[PAR_NB + 3 * 384],
        &sm_par[PAR_AG + 3 * 384],   &sm_par[PAR_AB + 3 * 384], nullptr);

    // ---- S3: gconvex(exF_n/exE_n) and half-gconv(IaiF_n/IaiE_n) ----
    for (int g = 0; g < 4; ++g) {
#pragma unroll
      for (int j = 0; j < 8; ++j) {
        int k = kg + 16 * j;
        frag_write(sm_hxfrag, m, k, sm_states[g * 2048 + m * 128 + k]);
      }
      __syncthreads();
      {
        int nt = wave;                          // 8 output tiles
        v8f acc = {0.f,0.f,0.f,0.f,0.f,0.f,0.f,0.f};
        int n = lane & 15, mb = (lane >> 4) * 8;
        if (g < 2) {                            // convex: full 128-wide output
          const unsigned short* bt = wcvex_all + g * 8192;
          int akt0 = (nt < 4) ? 0 : 2;
#pragma unroll
          for (int kt = 0; kt < 2; ++kt)
            acc = wmma_lg(sm_hxfrag + (akt0 + kt) * 512 + lane * 16,
                          bt + (kt * 8 + nt) * 512 + lane * 16, acc);
          float bias = sm_par[PAR_CVEXB + g * 128 + nt * 16 + n];
#pragma unroll
          for (int r = 0; r < 8; ++r)
            sm_g[g * 2048 + (mb + r) * 128 + nt * 16 + n] = acc[r] + bias;
        } else {                                // only c[:,128:] halves are used
          int l  = (g == 2) ? 4 : 5;            // convW[4]<-IaiF_n, convW[5]<-IaiE_n
          int gb = (g == 2) ? 3 : 2;            // sm_g[3]=IaiF2mnE, sm_g[2]=IaiE2mnF
          const unsigned short* bt = wconv_all + l * 16384;
          int ntg = 8 + nt;
#pragma unroll
          for (int kt = 0; kt < 2; ++kt)
            acc = wmma_lg(sm_hxfrag + (2 + kt) * 512 + lane * 16,
                          bt + (kt * 16 + ntg) * 512 + lane * 16, acc);
          float bias = sm_par[PAR_CONVB + l * 256 + ntg * 16 + n];
#pragma unroll
          for (int r = 0; r < 8; ++r)
            sm_g[gb * 2048 + (mb + r) * 128 + nt * 16 + n] = acc[r] + bias;
        }
      }
      __syncthreads();
    }

    // ---- mnF (li=4): xs = {IaF2mn, exF2mn, IaiE2mnF}; writes out cols 0..127
    run_layer<3, true>(tid, wg, t,
        &sm_cin[0 * 4096], 256, &sm_g[0 * 2048], 128, &sm_g[2 * 2048], 128,
        &sm_states[4 * 2048], sm_h2, sm_x, sm_hxfrag, sm_xfrag,
        wh2h_all + 4 * 32768, wx2h_all + 4 * 16384,
        &sm_par[PAR_H2LW + 4 * 256], &sm_par[PAR_H2LB + 4 * 256],
        &sm_par[PAR_XLW + 4 * 128],  &sm_par[PAR_XLB + 4 * 128],
        &sm_par[PAR_NG + 4 * 384],   &sm_par[PAR_NB + 4 * 384],
        &sm_par[PAR_AG + 4 * 384],   &sm_par[PAR_AB + 4 * 384],
        out + 0);
    // ---- mnE (li=5): xs = {IaE2mn, exE2mn, IaiF2mnE}; writes out cols 128..255
    run_layer<3, true>(tid, wg, t,
        &sm_cin[2 * 4096], 256, &sm_g[1 * 2048], 128, &sm_g[3 * 2048], 128,
        &sm_states[5 * 2048], sm_h2, sm_x, sm_hxfrag, sm_xfrag,
        wh2h_all + 5 * 32768, wx2h_all + 5 * 16384,
        &sm_par[PAR_H2LW + 5 * 256], &sm_par[PAR_H2LB + 5 * 256],
        &sm_par[PAR_XLW + 5 * 128],  &sm_par[PAR_XLB + 5 * 128],
        &sm_par[PAR_NG + 5 * 384],   &sm_par[PAR_NB + 5 * 384],
        &sm_par[PAR_AG + 5 * 384],   &sm_par[PAR_AB + 5 * 384],
        out + (size_t)128 * 256);
  }
}

// ---------------------------------------------------------------------------
extern "C" void kernel_launch(void* const* d_in, const int* in_sizes, int n_in,
                              void* d_out, int out_size, void* d_ws, size_t ws_size,
                              hipStream_t stream) {
  const float* aff      = (const float*)d_in[0];
  const float* norm_g   = (const float*)d_in[1];
  const float* norm_b   = (const float*)d_in[2];
  const float* aff_g    = (const float*)d_in[3];
  const float* aff_b    = (const float*)d_in[4];
  const float* x2h_W    = (const float*)d_in[5];
  const float* x2h_lnw  = (const float*)d_in[6];
  const float* x2h_lnb  = (const float*)d_in[7];
  const float* h2h_W    = (const float*)d_in[8];
  const float* h2h_lnw  = (const float*)d_in[9];
  const float* h2h_lnb  = (const float*)d_in[10];
  const float* convW    = (const float*)d_in[11];
  const float* convb    = (const float*)d_in[12];
  const float* convexW  = (const float*)d_in[13];
  const float* convexb  = (const float*)d_in[14];

  char* ws = (char*)d_ws;
  unsigned short* ws_h2h  = (unsigned short*)(ws + OFF_H2H);
  unsigned short* ws_x2h  = (unsigned short*)(ws + OFF_X2H);
  unsigned short* ws_conv = (unsigned short*)(ws + OFF_CONV);
  unsigned short* ws_cvex = (unsigned short*)(ws + OFF_CVEX);
  unsigned short* ws_aff  = (unsigned short*)(ws + OFF_AFF);

  // one-time (per launch) fragment packing
  pack_w<<<768, 256, 0, stream>>>(h2h_W,   ws_h2h,  128, 256, 6);
  pack_w<<<384, 256, 0, stream>>>(x2h_W,   ws_x2h,  128, 128, 6);
  pack_w<<<384, 256, 0, stream>>>(convW,   ws_conv,  64, 256, 6);
  pack_w<<< 64, 256, 0, stream>>>(convexW, ws_cvex,  64, 128, 2);
  pack_aff<<<131072, 256, 0, stream>>>(aff, ws_aff);

  spinal_scan<<<16, 256, 0, stream>>>(
      ws_aff, ws_h2h, ws_x2h, ws_conv, ws_cvex,
      norm_g, norm_b, aff_g, aff_b,
      x2h_lnw, x2h_lnb, h2h_lnw, h2h_lnb,
      convb, convexb, (float*)d_out);
}
